// IDynamicLayerBlock_28209345200512
// MI455X (gfx1250) — compile-verified, matrix operand order checked by reading, not run on hardware
//
#include <hip/hip_runtime.h>
#include <math.h>

// ---------------------------------------------------------------------------
// Types for CDNA5 WMMA (wave32, 16x16x32 bf16 -> f32)
// ---------------------------------------------------------------------------
typedef __attribute__((ext_vector_type(16))) __bf16       v16bf;
typedef __attribute__((ext_vector_type(8)))  float        v8f;
typedef __attribute__((ext_vector_type(4)))  unsigned int u32x4;

union FragBF { v16bf v; u32x4 q[2]; };

__device__ __forceinline__ float bf2f(unsigned short h) {
    unsigned int u = ((unsigned int)h) << 16;
    float f; __builtin_memcpy(&f, &u, 4); return f;
}
__device__ __forceinline__ unsigned short f2bf(float f) {
    unsigned int u; __builtin_memcpy(&u, &f, 4);
    u = (u + 0x7FFFu + ((u >> 16) & 1u)) >> 16;     // round-to-nearest-even
    return (unsigned short)u;
}

// Async global->LDS copy (CDNA5 LDS-DMA, tracked by ASYNCcnt).
// LDS operand = low 32 bits of the generic pointer;
// hardware adds the wave's LDS base.
__device__ __forceinline__ void async_ld_b128(void* lds_dst, const void* gsrc) {
    unsigned loff          = (unsigned)(size_t)lds_dst;
    unsigned long long gad = (unsigned long long)(size_t)gsrc;
    asm volatile("global_load_async_to_lds_b128 %0, %1, off"
                 :: "v"(loff), "v"(gad) : "memory");
}
__device__ __forceinline__ void wait_async0() {
    asm volatile("s_wait_asynccnt 0x0" ::: "memory");
}

#define DIM   180
#define CP    192        // padded channel dim (multiple of 32)
#define NPIX  32768      // 8 * 64 * 64
#define HID   360
#define H2    720

// ---------------------------------------------------------------------------
// Weight prep: fp32 [R x C] -> padded bf16 (or fp32) [Rp x Cp], zero pad.
// ---------------------------------------------------------------------------
__global__ __launch_bounds__(256) void pad_convert_kernel(
    const float* __restrict__ src, int R, int C, int Rp, int Cp,
    unsigned short* __restrict__ dstB, float* __restrict__ dstF) {
    int idx = blockIdx.x * 256 + threadIdx.x;
    if (idx >= Rp * Cp) return;
    int r = idx / Cp, c = idx - r * Cp;
    float v = (r < R && c < C) ? src[(size_t)r * C + c] : 0.f;
    if (dstB) dstB[idx] = f2bf(v);
    if (dstF) dstF[idx] = v;
}

// ---------------------------------------------------------------------------
// LN1: NCHW fp32 -> pixel-major bf16 [N][192] (LN applied) + fp32 raw copy.
// One block = 64 pixels of one image; LDS transpose tile 180x64 (stride 65).
// ---------------------------------------------------------------------------
__global__ __launch_bounds__(256) void ln1_kernel(
    const float* __restrict__ x, const float* __restrict__ lnw,
    const float* __restrict__ lnb,
    unsigned short* __restrict__ xn, float* __restrict__ xpm) {
    __shared__ float ldsX[DIM * 65];
    __shared__ float ldsS[256], ldsS2[256];
    __shared__ float muA[64], rsA[64];

    int tid = threadIdx.x;
    int b   = blockIdx.x >> 6;
    int p0  = (blockIdx.x & 63) * 64;
    int pl  = tid & 63;          // pixel lane
    int cq  = tid >> 6;          // channel quarter 0..3

    float s = 0.f, s2 = 0.f;
    #pragma unroll 5
    for (int i = 0; i < 45; ++i) {
        int c = cq + 4 * i;                       // covers 0..179
        float v = x[((size_t)(b * DIM + c)) * 4096 + p0 + pl];
        ldsX[c * 65 + pl] = v;
        s += v; s2 += v * v;
    }
    ldsS[cq * 64 + pl] = s;  ldsS2[cq * 64 + pl] = s2;
    __syncthreads();
    if (tid < 64) {
        float S = 0.f, S2 = 0.f;
        for (int q = 0; q < 4; ++q) { S += ldsS[q * 64 + tid]; S2 += ldsS2[q * 64 + tid]; }
        float m = S / (float)DIM;
        muA[tid] = m;
        rsA[tid] = rsqrtf(S2 / (float)DIM - m * m + 1e-5f);
    }
    __syncthreads();
    for (int idx = tid; idx < 64 * CP; idx += 256) {
        int p = idx / CP, c = idx - p * CP;
        size_t gp = (size_t)b * 4096 + p0 + p;
        float raw = (c < DIM) ? ldsX[c * 65 + p] : 0.f;
        float nv  = (c < DIM) ? (raw - muA[p]) * rsA[p] * lnw[c] + lnb[c] : 0.f;
        xn[gp * CP + c]  = f2bf(nv);
        xpm[gp * CP + c] = raw;
    }
}

// ---------------------------------------------------------------------------
// WMMA bf16 GEMM: C[M x N] = A[M x K] * B[N x K]^T (+bias) (+fp32 residual)
// Block tile 128x64, 8 waves (4x2), each wave 2x2 WMMA tiles (32x32).
// Double-buffered LDS staging via async global->LDS DMA (ASYNCcnt): the
// k+32 slab streams into buffer buf^1 while WMMAs consume buffer buf.
// LDS rows padded to 40 halfwords (80 B: 16B-aligned, conflict-free
// ds_load_b128 fragment reads in the ISA 16-bit A 16x32 lane layout).
// ---------------------------------------------------------------------------
__global__ __launch_bounds__(256) void wmma_gemm_kernel(
    const unsigned short* __restrict__ A, int lda,
    const unsigned short* __restrict__ B, int ldb,
    int M, int N, int K,
    const float* __restrict__ bias,
    const float* __restrict__ resid, int ldr,
    unsigned short* __restrict__ Cb, float* __restrict__ Cf, int ldc) {

    __shared__ unsigned short sA[2][128 * 40];
    __shared__ unsigned short sB[2][64 * 40];

    int tid    = threadIdx.x;
    int lane   = tid & 31;
    int wave   = tid >> 5;
    int wm     = wave >> 1;            // 0..3
    int wn     = wave & 1;             // 0..1
    int half   = lane >> 4;
    int li     = lane & 15;
    int blockM = blockIdx.x * 128;
    int blockN = blockIdx.y * 64;

    v8f acc[2][2];
    acc[0][0] = {}; acc[0][1] = {}; acc[1][0] = {}; acc[1][1] = {};

    // stage one 32-wide K slab into LDS buffer `buf` (3 async b128 per thread)
    auto stage = [&](int buf, int k0) {
        #pragma unroll
        for (int s = tid; s < 512; s += 256) {          // A: 128 rows x 4 segs
            int r = s >> 2, seg = s & 3;
            async_ld_b128(&sA[buf][r * 40 + seg * 8],
                          A + (size_t)(blockM + r) * lda + k0 + seg * 8);
        }
        {                                               // B: 64 rows x 4 segs
            int r = tid >> 2, seg = tid & 3;
            async_ld_b128(&sB[buf][r * 40 + seg * 8],
                          B + (size_t)(blockN + r) * ldb + k0 + seg * 8);
        }
    };

    stage(0, 0);
    wait_async0();
    __syncthreads();

    int buf = 0;
    for (int k0 = 0; k0 < K; k0 += 32) {
        if (k0 + 32 < K) stage(buf ^ 1, k0 + 32);   // overlap DMA with WMMA

        FragBF a[2], bfr[2];
        #pragma unroll
        for (int i = 0; i < 2; ++i) {
            int row = wm * 32 + i * 16 + li;
            a[i].q[0] = *(const u32x4*)(&sA[buf][row * 40 + half * 8]);      // K 0..7 / 8..15
            a[i].q[1] = *(const u32x4*)(&sA[buf][row * 40 + 16 + half * 8]); // K 16..23 / 24..31
        }
        #pragma unroll
        for (int j = 0; j < 2; ++j) {
            int col = wn * 32 + j * 16 + li;
            bfr[j].q[0] = *(const u32x4*)(&sB[buf][col * 40 + half * 8]);
            bfr[j].q[1] = *(const u32x4*)(&sB[buf][col * 40 + 16 + half * 8]);
        }
        #pragma unroll
        for (int i = 0; i < 2; ++i)
            #pragma unroll
            for (int j = 0; j < 2; ++j)
                acc[i][j] = __builtin_amdgcn_wmma_f32_16x16x32_bf16(
                    false, a[i].v, false, bfr[j].v, (short)0, acc[i][j], false, false);

        wait_async0();        // next-buffer DMA complete for this wave
        __syncthreads();      // all waves done reading buf + done staging
        buf ^= 1;
    }

    // epilogue: lane holds col = li, rows half*8 + r  (per ISA C/D layout)
    #pragma unroll
    for (int i = 0; i < 2; ++i) {
        #pragma unroll
        for (int j = 0; j < 2; ++j) {
            int col = blockN + wn * 32 + j * 16 + li;
            if (col >= N) continue;
            float bia = bias ? bias[col] : 0.f;
            #pragma unroll
            for (int r = 0; r < 8; ++r) {
                int row = blockM + wm * 32 + i * 16 + half * 8 + r;
                float v = acc[i][j][r] + bia;
                if (resid) v += resid[(size_t)row * ldr + col];
                if (Cf) Cf[(size_t)row * ldc + col] = v;
                if (Cb) Cb[(size_t)row * ldc + col] = f2bf(v);
            }
        }
    }
}

// ---------------------------------------------------------------------------
// Involution apply: out[p][c] = sum_{7x7} WGT[p][(c/6)*49+tap] * Y1[p+d][c]
// 16x16 pixel tile per block, 22x22 halo of Y1 staged in LDS in 3 channel
// chunks of 60 (LDS stride 62 -> conflict-free).
// ---------------------------------------------------------------------------
__global__ __launch_bounds__(256) void idyn_kernel(
    const unsigned short* __restrict__ Y1, const unsigned short* __restrict__ WGT,
    unsigned short* __restrict__ out) {
    __shared__ unsigned short ldsY[484 * 62];

    int tid  = threadIdx.x;
    int b    = blockIdx.x >> 4;
    int tile = blockIdx.x & 15;
    int ty0  = (tile >> 2) * 16;
    int tx0  = (tile & 3) * 16;
    int ty   = tid >> 4, tx = tid & 15;

    size_t pix = (size_t)b * 4096 + (ty0 + ty) * 64 + (tx0 + tx);
    const unsigned short* wrow = WGT + pix * 1472;

    for (int chunk = 0; chunk < 3; ++chunk) {
        int c0 = chunk * 60;
        // load 22x22 halo, 60 channels (30 dwords per pixel), zero OOB
        for (int idx = tid; idx < 484 * 30; idx += 256) {
            int pixL = idx / 30, d = idx - pixL * 30;
            int hy = pixL / 22, hx = pixL - hy * 22;
            int gy = ty0 + hy - 3, gx = tx0 + hx - 3;
            unsigned int val = 0u;
            if (gy >= 0 && gy < 64 && gx >= 0 && gx < 64)
                val = *(const unsigned int*)(Y1 + ((size_t)b * 4096 + gy * 64 + gx) * CP + c0 + d * 2);
            *(unsigned int*)(&ldsY[pixL * 62 + d * 2]) = val;
        }
        __syncthreads();

        for (int g = 0; g < 10; ++g) {                 // 10 groups x 6 ch = 60
            float w[49];
            #pragma unroll
            for (int q = 0; q < 49; ++q)
                w[q] = bf2f(wrow[(chunk * 10 + g) * 49 + q]);
            #pragma unroll
            for (int cc = 0; cc < 6; ++cc) {
                int cl = g * 6 + cc;
                float accv = 0.f;
                #pragma unroll
                for (int i = 0; i < 7; ++i) {
                    const unsigned short* rowp = &ldsY[((ty + i) * 22 + tx) * 62 + cl];
                    #pragma unroll
                    for (int j = 0; j < 7; ++j)
                        accv += w[i * 7 + j] * bf2f(rowp[j * 62]);
                }
                out[pix * CP + c0 + cl] = f2bf(accv);
            }
        }
        __syncthreads();
    }
    #pragma unroll
    for (int c = DIM; c < CP; ++c) out[pix * CP + c] = 0;   // K-pad for next GEMM
}

// ---------------------------------------------------------------------------
// LN2 on pixel-major fp32 rows (wave per pixel, wave32 shfl reduction).
// ---------------------------------------------------------------------------
__global__ __launch_bounds__(256) void ln2_kernel(
    const float* __restrict__ x1, const float* __restrict__ lnw,
    const float* __restrict__ lnb, unsigned short* __restrict__ out) {
    int lane = threadIdx.x & 31;
    size_t pix = (size_t)blockIdx.x * 8 + (threadIdx.x >> 5);
    const float* row = x1 + pix * CP;

    float v[6], s = 0.f, s2 = 0.f;
    #pragma unroll
    for (int k = 0; k < 6; ++k) {
        int c = lane + 32 * k;
        float t = (c < DIM) ? row[c] : 0.f;
        v[k] = t; s += t; s2 += t * t;
    }
    #pragma unroll
    for (int off = 16; off > 0; off >>= 1) {
        s  += __shfl_xor(s,  off, 32);
        s2 += __shfl_xor(s2, off, 32);
    }
    float mu = s / (float)DIM;
    float rs = rsqrtf(s2 / (float)DIM - mu * mu + 1e-5f);
    #pragma unroll
    for (int k = 0; k < 6; ++k) {
        int c = lane + 32 * k;
        unsigned short o = 0;
        if (c < DIM) o = f2bf((v[k] - mu) * rs * lnw[c] + lnb[c]);
        out[pix * CP + c] = o;                         // c up to 191: zero pad
    }
}

// ---------------------------------------------------------------------------
// GDFN: depthwise 3x3 on Z[N][720], exact-GELU gate -> gated[N][384] bf16.
// ---------------------------------------------------------------------------
__global__ __launch_bounds__(256) void gdfn_dw_kernel(
    const unsigned short* __restrict__ Z, const float* __restrict__ dww,
    unsigned short* __restrict__ gated) {
    size_t idx = (size_t)blockIdx.x * 256 + threadIdx.x;
    if (idx >= (size_t)NPIX * 384) return;
    size_t p = idx / 384;
    int c = (int)(idx - p * 384);
    if (c >= HID) { gated[idx] = 0; return; }          // K-pad for pout GEMM
    int b = (int)(p >> 12), pp = (int)(p & 4095);
    int hy = pp >> 6, hx = pp & 63;

    float s1 = 0.f, s2 = 0.f;
    #pragma unroll
    for (int i = -1; i <= 1; ++i) {
        #pragma unroll
        for (int j = -1; j <= 1; ++j) {
            int yy = hy + i, xx = hx + j;
            if (yy < 0 || yy >= 64 || xx < 0 || xx >= 64) continue;
            const unsigned short* zp =
                Z + ((size_t)b * 4096 + yy * 64 + xx) * H2 + c;
            int tap = (i + 1) * 3 + (j + 1);
            s1 += dww[c * 9 + tap]          * bf2f(zp[0]);
            s2 += dww[(c + HID) * 9 + tap]  * bf2f(zp[HID]);
        }
    }
    float g = 0.5f * s1 * (1.f + erff(s1 * 0.70710678118f));
    gated[idx] = f2bf(g * s2);
}

// ---------------------------------------------------------------------------
// Pixel-major fp32 [N][192] -> NCHW fp32 output, LDS 32x32 transpose tiles.
// ---------------------------------------------------------------------------
__global__ __launch_bounds__(256) void to_nchw_kernel(
    const float* __restrict__ res, float* __restrict__ out) {
    __shared__ float tile[32][33];
    int tid = threadIdx.x;
    int ct = blockIdx.x % 6;
    int rest = blockIdx.x / 6;
    int pt = rest & 127;
    int b  = rest >> 7;
    int c0 = ct * 32, p0 = pt * 32;

    for (int s = tid; s < 1024; s += 256) {
        int pi = s >> 5, cj = s & 31;
        int c = c0 + cj;
        float v = (c < DIM) ? res[((size_t)b * 4096 + p0 + pi) * CP + c] : 0.f;
        tile[cj][pi] = v;
    }
    __syncthreads();
    for (int s = tid; s < 1024; s += 256) {
        int cj = s >> 5, pi = s & 31;
        int c = c0 + cj;
        if (c < DIM)
            out[((size_t)(b * DIM + c)) * 4096 + p0 + pi] = tile[cj][pi];
    }
}

// ---------------------------------------------------------------------------
// Host orchestration
// ---------------------------------------------------------------------------
static inline void launch_gemm(hipStream_t st,
                               const unsigned short* A, int lda,
                               const unsigned short* B, int ldb,
                               int M, int N, int K,
                               const float* bias, const float* resid, int ldr,
                               unsigned short* Cb, float* Cf, int ldc) {
    dim3 grid(M / 128, (N + 63) / 64);
    wmma_gemm_kernel<<<grid, 256, 0, st>>>(A, lda, B, ldb, M, N, K,
                                           bias, resid, ldr, Cb, Cf, ldc);
}

extern "C" void kernel_launch(void* const* d_in, const int* in_sizes, int n_in,
                              void* d_out, int out_size, void* d_ws, size_t ws_size,
                              hipStream_t stream) {
    const float* x       = (const float*)d_in[0];
    const float* ln1_w   = (const float*)d_in[1];
    const float* ln1_b   = (const float*)d_in[2];
    const float* conv0_w = (const float*)d_in[3];
    const float* dg_w1   = (const float*)d_in[4];
    const float* dg_b1   = (const float*)d_in[5];
    const float* dg_w2   = (const float*)d_in[6];
    const float* dg_b2   = (const float*)d_in[7];
    const float* conv1_w = (const float*)d_in[8];
    const float* ln2_w   = (const float*)d_in[9];
    const float* ln2_b   = (const float*)d_in[10];
    const float* pin_w   = (const float*)d_in[11];
    const float* dw_w    = (const float*)d_in[12];
    const float* pout_w  = (const float*)d_in[13];
    float* out = (float*)d_out;

    // workspace arena (peak ~190 MB, buffers reused)
    char* ws = (char*)d_ws;
    size_t off = 0;
    auto give = [&](size_t bytes) -> char* {
        char* p = ws + off; off = (off + bytes + 255) & ~(size_t)255; return p;
    };
    unsigned short* Xn   = (unsigned short*)give((size_t)NPIX * CP * 2);   // LN1 out; reused for LN2 out
    float*          xpm  = (float*)         give((size_t)NPIX * CP * 4);   // raw x pm; reused for final result
    unsigned short* Y1   = (unsigned short*)give((size_t)NPIX * CP * 2);
    unsigned short* T    = (unsigned short*)give((size_t)NPIX * 64 * 2);
    char*           aren = give((size_t)NPIX * 1472 * 2);                  // WGT; reused for Z + gated
    unsigned short* WGT  = (unsigned short*)aren;
    unsigned short* Zb   = (unsigned short*)aren;
    unsigned short* gat  = (unsigned short*)(aren + (size_t)NPIX * H2 * 2);
    unsigned short* idyn = (unsigned short*)give((size_t)NPIX * CP * 2);
    float*          x1   = (float*)         give((size_t)NPIX * CP * 4);
    // padded bf16 weights + fp32 biases
    unsigned short* w0b  = (unsigned short*)give(192 * 192 * 2);
    unsigned short* w1gb = (unsigned short*)give(64 * 192 * 2);
    unsigned short* w2gb = (unsigned short*)give(1472 * 64 * 2);
    unsigned short* w1b  = (unsigned short*)give(192 * 192 * 2);
    unsigned short* pinb = (unsigned short*)give(768 * 192 * 2);
    unsigned short* potb = (unsigned short*)give(192 * 384 * 2);
    float*          b1p  = (float*)give(64 * 4);
    float*          b2p  = (float*)give(1472 * 4);
    float*          res  = xpm;                                            // reuse
    unsigned short* Xn2  = Xn;                                             // reuse

    auto pad = [&](const float* s, int R, int C, int Rp, int Cp,
                   unsigned short* dB, float* dF) {
        int tot = Rp * Cp;
        pad_convert_kernel<<<(tot + 255) / 256, 256, 0, stream>>>(s, R, C, Rp, Cp, dB, dF);
    };
    pad(conv0_w, 180, 180, 192, 192, w0b, nullptr);
    pad(dg_w1,   45,  180, 64,  192, w1gb, nullptr);
    pad(dg_w2,   1470, 45, 1472, 64, w2gb, nullptr);
    pad(conv1_w, 180, 180, 192, 192, w1b, nullptr);
    pad(pin_w,   720, 180, 768, 192, pinb, nullptr);
    pad(pout_w,  180, 360, 192, 384, potb, nullptr);
    pad(dg_b1,   45, 1, 64, 1, nullptr, b1p);
    pad(dg_b2,   1470, 1, 1472, 1, nullptr, b2p);

    // branch 1
    ln1_kernel<<<512, 256, 0, stream>>>(x, ln1_w, ln1_b, Xn, xpm);
    launch_gemm(stream, Xn, CP, w0b, CP, NPIX, 192, 192,
                nullptr, nullptr, 0, Y1, nullptr, CP);                     // conv0
    launch_gemm(stream, Y1, CP, w1gb, CP, NPIX, 64, 192,
                b1p, nullptr, 0, T, nullptr, 64);                          // dg_w1 + b1
    launch_gemm(stream, T, 64, w2gb, 64, NPIX, 1472, 64,
                b2p, nullptr, 0, WGT, nullptr, 1472);                      // dg_w2 + b2
    idyn_kernel<<<128, 256, 0, stream>>>(Y1, WGT, idyn);                   // involution
    launch_gemm(stream, idyn, CP, w1b, CP, NPIX, 192, 192,
                nullptr, xpm, CP, nullptr, x1, CP);                        // conv1 + residual -> x1

    // branch 2
    ln2_kernel<<<NPIX / 8, 256, 0, stream>>>(x1, ln2_w, ln2_b, Xn2);
    launch_gemm(stream, Xn2, CP, pinb, CP, NPIX, 720, 192,
                nullptr, nullptr, 0, Zb, nullptr, H2);                     // pin
    {
        size_t tot = (size_t)NPIX * 384;
        gdfn_dw_kernel<<<(unsigned)((tot + 255) / 256), 256, 0, stream>>>(Zb, dw_w, gat);
    }
    launch_gemm(stream, gat, 384, potb, 384, NPIX, 192, 384,
                nullptr, x1, CP, nullptr, res, CP);                        // pout + residual

    to_nchw_kernel<<<8 * 128 * 6, 256, 0, stream>>>(res, out);
}
